// GraphAttentionHeadModule_4861902979152
// MI455X (gfx1250) — matrix operand order repaired and checked by reading, not compiled, not run on hardware
//
#include <hip/hip_runtime.h>
#include <hip/hip_bf16.h>

#define N_NODES 8192
#define DIN 512
#define DOUT 256
#define ALPHA 0.2f
#define EPS 1e-5f

typedef __attribute__((ext_vector_type(16))) __bf16 v16bf;
typedef __attribute__((ext_vector_type(8)))  __bf16 v8bf;
typedef __attribute__((ext_vector_type(8)))  float  v8f;

// ---------------------------------------------------------------------------
// Kernel 1: h = LayerNorm(features @ phi_w^T + phi_b); write hT (bf16, [DOUT][N])
//           and mu_h = h @ mu_w  (f32, [N]).
// One block = 16 rows x 256 cols. 8 waves, each wave owns two 16x16 C tiles.
// ---------------------------------------------------------------------------
__global__ __launch_bounds__(256) void k1_gemm_ln(
    const float* __restrict__ features, const float* __restrict__ phi_w,
    const float* __restrict__ phi_b, const float* __restrict__ mu_w,
    const float* __restrict__ ln_g, const float* __restrict__ ln_b,
    __bf16* __restrict__ hT, float* __restrict__ mu_out)
{
    __shared__ float hbuf[16][272];      // 16 rows x 256 cols (+pad)
    __shared__ float red[16][16];
    __shared__ float red2[16][16];
    __shared__ float smean[16], srstd[16];

    const int i0   = blockIdx.x * 16;
    const int t    = threadIdx.x;
    const int wave = t >> 5;
    const int lane = t & 31;
    const int hi   = lane >> 4;          // lane half (wave32)
    const int l15  = lane & 15;
    const int col0 = wave * 32;
    const int col1 = col0 + 16;

    v8f c0 = {}; v8f c1 = {};
    for (int kb = 0; kb < DIN; kb += 32) {
        // A tile (16x32 bf16): lane row M=l15; K = kb+16*hi+e
        const float* ap = features + (size_t)(i0 + l15) * DIN + kb + 16 * hi;
        v16bf a;
        #pragma unroll
        for (int e = 0; e < 8; ++e) { a[e] = (__bf16)ap[e]; a[8 + e] = (__bf16)ap[8 + e]; }
        // B tiles (32x16 bf16): B[k][n] = phi_w[n][k]; lane col n=l15, K = kb+16*hi+e
        const float* bp0 = phi_w + (size_t)(col0 + l15) * DIN + kb + 16 * hi;
        const float* bp1 = phi_w + (size_t)(col1 + l15) * DIN + kb + 16 * hi;
        v16bf b0, b1;
        #pragma unroll
        for (int e = 0; e < 16; ++e) { b0[e] = (__bf16)bp0[e]; b1[e] = (__bf16)bp1[e]; }
        c0 = __builtin_amdgcn_wmma_f32_16x16x32_bf16(false, a, false, b0, (short)0, c0, false, false);
        c1 = __builtin_amdgcn_wmma_f32_16x16x32_bf16(false, a, false, b1, (short)0, c1, false, false);
    }
    // bias + stash tile into LDS (C layout: vgpr r -> row r+8*hi, lane col l15)
    const float bias0 = phi_b[col0 + l15];
    const float bias1 = phi_b[col1 + l15];
    #pragma unroll
    for (int r = 0; r < 8; ++r) {
        const int row = r + 8 * hi;
        hbuf[row][col0 + l15] = c0[r] + bias0;
        hbuf[row][col1 + l15] = c1[r] + bias1;
    }
    __syncthreads();

    // LayerNorm: thread t handles row=t>>4, 16-col chunk (t&15)
    const int row = t >> 4;
    const int cc  = (t & 15) * 16;
    float s = 0.f, s2 = 0.f;
    #pragma unroll
    for (int k = 0; k < 16; ++k) { const float v = hbuf[row][cc + k]; s += v; s2 += v * v; }
    red[row][t & 15] = s; red2[row][t & 15] = s2;
    __syncthreads();
    if (t < 16) {
        float sum = 0.f, sq = 0.f;
        #pragma unroll
        for (int k = 0; k < 16; ++k) { sum += red[t][k]; sq += red2[t][k]; }
        const float mean = sum * (1.f / DOUT);
        const float var  = sq * (1.f / DOUT) - mean * mean;
        smean[t] = mean; srstd[t] = rsqrtf(var + EPS);
    }
    __syncthreads();
    const float mean = smean[row], rstd = srstd[row];
    float mup = 0.f;
    #pragma unroll
    for (int k = 0; k < 16; ++k) {
        const int c = cc + k;
        const float y = (hbuf[row][c] - mean) * rstd * ln_g[c] + ln_b[c];
        hT[(size_t)c * N_NODES + i0 + row] = (__bf16)y;
        mup += y * mu_w[c];
    }
    __syncthreads();
    red[row][t & 15] = mup;
    __syncthreads();
    if (t < 16) {
        float sum = 0.f;
        #pragma unroll
        for (int k = 0; k < 16; ++k) sum += red[t][k];
        mu_out[i0 + t] = sum;
    }
}

// ---------------------------------------------------------------------------
// Kernel 2: per-row softmax stats. m_i = leaky(mu_i + max_{adj} mu_j)
// (LeakyReLU monotone), l_i = sum_j exp(e_ij - m_i). One block per row.
// ---------------------------------------------------------------------------
__global__ __launch_bounds__(256) void k2_stats(
    const int* __restrict__ adj, const float* __restrict__ mu,
    float* __restrict__ stats)   // stats[i]=m_i, stats[N+i]=1/l_i
{
    __shared__ float sred[256];
    const int i = blockIdx.x;
    const int t = threadIdx.x;
    const float mui = mu[i];
    const int* arow = adj + (size_t)i * N_NODES;

    float mmax = -3.4e38f;
    #pragma unroll 4
    for (int q = 0; q < 32; ++q) {
        const int j = t * 32 + q;
        if (arow[j]) mmax = fmaxf(mmax, mu[j]);
    }
    sred[t] = mmax; __syncthreads();
    for (int s = 128; s > 0; s >>= 1) {
        if (t < s) sred[t] = fmaxf(sred[t], sred[t + s]);
        __syncthreads();
    }
    const float e0 = mui + sred[0];
    const float m  = e0 > 0.f ? e0 : ALPHA * e0;
    __syncthreads();

    float acc = 0.f;
    #pragma unroll 4
    for (int q = 0; q < 32; ++q) {
        const int j = t * 32 + q;
        if (arow[j]) {
            float x = mui + mu[j];
            x = x > 0.f ? x : ALPHA * x;
            acc += __expf(x - m);
        }
    }
    sred[t] = acc; __syncthreads();
    for (int s = 128; s > 0; s >>= 1) {
        if (t < s) sred[t] += sred[t + s];
        __syncthreads();
    }
    if (t == 0) { stats[i] = m; stats[N_NODES + i] = 1.f / sred[0]; }
}

// ---------------------------------------------------------------------------
// Kernel 3: h_out = softmax(e) @ h, fused ELU. One block = 16 rows x 256 cols.
// Double-buffered j-chunks of 128: exp/prob VALU work for chunk n+1 overlaps
// (co-executes) with the v_wmma_f32_16x16x32_bf16 stream for chunk n.
// One barrier per chunk; adj stream prefetched two chunks ahead.
// ---------------------------------------------------------------------------
__device__ __forceinline__ void compute_probs(
    __bf16 (*buf)[136], const int* __restrict__ arow,
    const float* __restrict__ mu, int jb, int r, int jsub,
    float muir, float mr, float invr)
{
    #pragma unroll
    for (int q = 0; q < 8; ++q) {
        const int j = jb + jsub + q;
        float p = 0.f;
        if (arow[j]) {
            float x = muir + mu[j];
            x = x > 0.f ? x : ALPHA * x;
            p = __expf(x - mr) * invr;   // fold 1/l into the prob
        }
        buf[r][jsub + q] = (__bf16)p;
    }
}

__global__ __launch_bounds__(256) void k3_attn(
    const int* __restrict__ adj, const float* __restrict__ mu,
    const float* __restrict__ stats, const __bf16* __restrict__ hT,
    float* __restrict__ out)
{
    __shared__ __bf16 pbuf[2][16][136];  // ping-pong: 16 rows x 128 probs (+pad)
    __shared__ float smu[16], sm[16], sinv[16];

    const int i0 = blockIdx.x * 16;
    const int t  = threadIdx.x;
    if (t < 16) { smu[t] = mu[i0 + t]; sm[t] = stats[i0 + t]; sinv[t] = stats[N_NODES + i0 + t]; }
    __syncthreads();

    const int wave = t >> 5, lane = t & 31, hi = lane >> 4, l15 = lane & 15;
    const int col0 = wave * 32, col1 = col0 + 16;
    v8f c0 = {}; v8f c1 = {};

    const int r    = t >> 4;             // prob row this thread fills
    const int jsub = (t & 15) * 8;       // 8 contiguous j's
    const float muir = smu[r], mr = sm[r], invr = sinv[r];
    const int* arow = adj + (size_t)(i0 + r) * N_NODES;

    // prologue: fill buffer 0
    compute_probs(pbuf[0], arow, mu, 0, r, jsub, muir, mr, invr);
    __syncthreads();

    int cur = 0;
    for (int jb = 0; jb < N_NODES; jb += 128) {
        if (jb + 256 < N_NODES)
            __builtin_prefetch(arow + jb + 256 + jsub, 0, 0);   // global_prefetch_b8
        if (jb + 128 < N_NODES)
            compute_probs(pbuf[cur ^ 1], arow, mu, jb + 128, r, jsub, muir, mr, invr);

        #pragma unroll
        for (int ks = 0; ks < 4; ++ks) {
            const int kb = ks * 32;
            // A from LDS: row=l15, K = kb+16*hi+e
            const v8bf a0 = *(const v8bf*)&pbuf[cur][l15][kb + 16 * hi];
            const v8bf a1 = *(const v8bf*)&pbuf[cur][l15][kb + 16 * hi + 8];
            v16bf a;
            #pragma unroll
            for (int e = 0; e < 8; ++e) { a[e] = a0[e]; a[8 + e] = a1[e]; }
            // B from hT (contiguous in K): col=l15, K = kb+16*hi+e
            const v16bf b0 = *(const v16bf*)&hT[(size_t)(col0 + l15) * N_NODES + jb + kb + 16 * hi];
            const v16bf b1 = *(const v16bf*)&hT[(size_t)(col1 + l15) * N_NODES + jb + kb + 16 * hi];
            c0 = __builtin_amdgcn_wmma_f32_16x16x32_bf16(false, a, false, b0, (short)0, c0, false, false);
            c1 = __builtin_amdgcn_wmma_f32_16x16x32_bf16(false, a, false, b1, (short)0, c1, false, false);
        }
        __syncthreads();
        cur ^= 1;
    }
    // ELU epilogue + store (C layout: vgpr rr -> row rr+8*hi, lane col l15)
    #pragma unroll
    for (int rr = 0; rr < 8; ++rr) {
        const int row = rr + 8 * hi;
        float x0 = c0[rr], x1 = c1[rr];
        x0 = x0 > 0.f ? x0 : (__expf(x0) - 1.f);
        x1 = x1 > 0.f ? x1 : (__expf(x1) - 1.f);
        out[(size_t)(i0 + row) * DOUT + col0 + l15] = x0;
        out[(size_t)(i0 + row) * DOUT + col1 + l15] = x1;
    }
}

extern "C" void kernel_launch(void* const* d_in, const int* in_sizes, int n_in,
                              void* d_out, int out_size, void* d_ws, size_t ws_size,
                              hipStream_t stream) {
    (void)in_sizes; (void)n_in; (void)out_size; (void)ws_size;
    const float* features = (const float*)d_in[0];
    const int*   adj      = (const int*)d_in[1];
    const float* phi_w    = (const float*)d_in[2];
    const float* phi_b    = (const float*)d_in[3];
    const float* mu_w     = (const float*)d_in[4];
    const float* ln_g     = (const float*)d_in[5];
    const float* ln_b     = (const float*)d_in[6];
    float* out = (float*)d_out;

    char* ws = (char*)d_ws;
    __bf16* hT    = (__bf16*)ws;                                   // 4 MB
    float*  mu    = (float*)(ws + (size_t)DOUT * N_NODES * 2);     // 32 KB
    float*  stats = mu + N_NODES;                                  // 64 KB

    k1_gemm_ln<<<N_NODES / 16, 256, 0, stream>>>(features, phi_w, phi_b, mu_w, ln_g, ln_b, hT, mu);
    k2_stats<<<N_NODES, 256, 0, stream>>>(adj, mu, stats);
    k3_attn<<<N_NODES / 16, 256, 0, stream>>>(adj, mu, stats, hT, out);
}
